// POMOForConvexHull_3195455668446
// MI455X (gfx1250) — matrix operand order repaired and checked by reading, not compiled, not run on hardware
//
#include <hip/hip_runtime.h>

// ---------------------------------------------------------------------------
// Problem constants (from reference)
// ---------------------------------------------------------------------------
#define NBATCH 64
#define NNODE  200
#define TOK    (NBATCH * NNODE)   // 12800 tokens
#define HID    128
#define NHEAD  8
#define DHEAD  16
#define DFF    2048
#define NLAYER 6
#define INDIM  26
#define QDIM   (3 * HID)          // 384
#define NEGV   (-1000000000.0f)
#define CSCALE 0.08838834764831845f   // 1/sqrt(128)

typedef __bf16 bf16;
typedef __attribute__((ext_vector_type(16))) __bf16 v16bf;
typedef __attribute__((ext_vector_type(8)))  __bf16 v8bf;
typedef __attribute__((ext_vector_type(8)))  float  v8f;

__device__ __forceinline__ bf16 f2bf(float f) {
  unsigned int u = __builtin_bit_cast(unsigned int, f);
  u += 0x7FFFu + ((u >> 16) & 1u);                 // round-to-nearest-even
  unsigned short h = (unsigned short)(u >> 16);
  return __builtin_bit_cast(bf16, h);
}

// A-fragment (16x32 bf16): contiguous 16-K chunk per lane half
// elements 0..15 <-> K = k0 + hi*16 + j
__device__ __forceinline__ v16bf load_afrag(const bf16* __restrict__ row, int k0, int kha) {
  return *(const v16bf*)(row + k0 + kha);
}

// B-fragment (32x16 bf16, stored as N x K): contiguous 16-K chunk per lane half
// elements 0..15 <-> K = k0 + hi*16 + j   -> one 32-byte load
__device__ __forceinline__ v16bf load_bfrag(const bf16* __restrict__ row, int k0, int khb) {
  return *(const v16bf*)(row + k0 + khb);
}

// ---------------------------------------------------------------------------
// WMMA GEMM:  C[M,N] = alpha * A[M,K] * B[N,K]^T + bias[N]   (bf16 in, f32 acc)
// One wave32 owns a 32x32 C tile: 2 A-frags x 2 B-frags -> 4 accumulators,
// 4 independent v_wmma_f32_16x16x32_bf16 per K-step (hides WMMA RAW latency,
// doubles FLOP/byte vs a single 16x16 tile). Block = 4 waves = 64x64 C tile.
// Optional batch via blockIdx.z strides. Edge tiles clamp loads / guard stores.
// ---------------------------------------------------------------------------
__global__ void wmma_gemm_kernel(const bf16* __restrict__ A, const bf16* __restrict__ B,
                                 const float* __restrict__ bias,
                                 float* __restrict__ Cf, bf16* __restrict__ Cb,
                                 int M, int N, int K, float alpha, int relu,
                                 long long sA, long long sB, long long sC)
{
  A += (long long)blockIdx.z * sA;
  B += (long long)blockIdx.z * sB;
  const long long cbase = (long long)blockIdx.z * sC;

  const int lane = threadIdx.x & 31;
  const int wave = threadIdx.x >> 5;               // 0..3, arranged 2x2
  const int m0 = (blockIdx.y * 2 + (wave >> 1)) * 32;
  const int n0 = (blockIdx.x * 2 + (wave & 1)) * 32;
  if (m0 >= M || n0 >= N) return;                  // wave-uniform

  const int r   = lane & 15;
  const int hi  = lane >> 4;                       // 0 or 1 (lane half)
  const int kha = hi * 16;
  const int khb = hi * 16;

  const int am0 = m0 + r,      am1 = m0 + 16 + r;
  const int bn0 = n0 + r,      bn1 = n0 + 16 + r;
  const bf16* A0 = A + (long long)(am0 < M ? am0 : 0) * K;
  const bf16* A1 = A + (long long)(am1 < M ? am1 : 0) * K;
  const bf16* B0 = B + (long long)(bn0 < N ? bn0 : 0) * K;
  const bf16* B1 = B + (long long)(bn1 < N ? bn1 : 0) * K;

  v8f acc00 = {}, acc01 = {}, acc10 = {}, acc11 = {};
  for (int k0 = 0; k0 < K; k0 += 32) {
    v16bf a0 = load_afrag(A0, k0, kha);
    v16bf a1 = load_afrag(A1, k0, kha);
    v16bf b0 = load_bfrag(B0, k0, khb);
    v16bf b1 = load_bfrag(B1, k0, khb);
    acc00 = __builtin_amdgcn_wmma_f32_16x16x32_bf16(false, a0, false, b0, (short)0, acc00, false, false);
    acc01 = __builtin_amdgcn_wmma_f32_16x16x32_bf16(false, a0, false, b1, (short)0, acc01, false, false);
    acc10 = __builtin_amdgcn_wmma_f32_16x16x32_bf16(false, a1, false, b0, (short)0, acc10, false, false);
    acc11 = __builtin_amdgcn_wmma_f32_16x16x32_bf16(false, a1, false, b1, (short)0, acc11, false, false);
  }

  // C/D layout: lane half hi covers rows +hi*8; column = lane&15
  const int col = lane & 15;
  v8f accs[2][2] = { { acc00, acc01 }, { acc10, acc11 } };
#pragma unroll
  for (int ti = 0; ti < 2; ++ti) {
#pragma unroll
    for (int tj = 0; tj < 2; ++tj) {
      const int cn = n0 + tj * 16 + col;
      if (cn >= N) continue;
      const float bb = bias ? bias[cn] : 0.0f;
      const int mb = m0 + ti * 16 + hi * 8;
#pragma unroll
      for (int i = 0; i < 8; ++i) {
        const int cm = mb + i;
        if (cm < M) {
          float v = accs[ti][tj][i] * alpha + bb;
          if (relu) v = fmaxf(v, 0.0f);
          const long long ci = cbase + (long long)cm * N + cn;
          if (Cf) Cf[ci] = v;
          if (Cb) Cb[ci] = f2bf(v);
        }
      }
    }
  }
}

// ---------------------------------------------------------------------------
// Elementwise f32 -> bf16 convert
// ---------------------------------------------------------------------------
__global__ void cvt_bf16_kernel(const float* __restrict__ s, bf16* __restrict__ d, long long n)
{
  long long i = (long long)blockIdx.x * blockDim.x + threadIdx.x;
  if (i < n) d[i] = f2bf(s[i]);
}

// ---------------------------------------------------------------------------
// Embedding: h = x @ embed_w^T + embed_b   (K=26, scalar)
// ---------------------------------------------------------------------------
__global__ void embed_kernel(const float* __restrict__ x, const float* __restrict__ w,
                             const float* __restrict__ b, float* __restrict__ h)
{
  int idx = blockIdx.x * blockDim.x + threadIdx.x;
  if (idx >= TOK * HID) return;
  int t = idx / HID, o = idx % HID;
  const float* xr = x + (long long)t * INDIM;
  const float* wr = w + (long long)o * INDIM;
  float s = b[o];
#pragma unroll
  for (int i = 0; i < INDIM; ++i) s += xr[i] * wr[i];
  h[idx] = s;
}

// ---------------------------------------------------------------------------
// Attention, flash-style online softmax. One thread per (batch, head, query).
// qkv layout: [tok][384] with q|k|v slices. Writes bf16 attn output [tok][128].
// ---------------------------------------------------------------------------
__global__ void attn_kernel(const float* __restrict__ qkv, bf16* __restrict__ ob)
{
  int idx = blockIdx.x * blockDim.x + threadIdx.x;
  if (idx >= NBATCH * NHEAD * NNODE) return;
  int qi = idx % NNODE; int tmp = idx / NNODE;
  int head = tmp % NHEAD; int b = tmp / NHEAD;

  const float* base = qkv + (long long)b * NNODE * QDIM;
  const float* qp = base + (long long)qi * QDIM + head * DHEAD;
  float q[DHEAD];
#pragma unroll
  for (int d = 0; d < DHEAD; ++d) q[d] = qp[d];

  float m = -3.4e38f, l = 0.0f, acc[DHEAD] = {};
  for (int n = 0; n < NNODE; ++n) {
    const float* kp = base + (long long)n * QDIM + HID + head * DHEAD;
    float s = 0.0f;
#pragma unroll
    for (int d = 0; d < DHEAD; ++d) s += q[d] * kp[d];
    s *= 0.25f;                                  // 1/sqrt(16)
    float nm = fmaxf(m, s);
    float c = __expf(m - nm), e = __expf(s - nm);
    l = l * c + e;
    const float* vp = kp + HID;
#pragma unroll
    for (int d = 0; d < DHEAD; ++d) acc[d] = acc[d] * c + e * vp[d];
    m = nm;
  }
  float inv = 1.0f / l;
  bf16* op = ob + (long long)(b * NNODE + qi) * HID + head * DHEAD;
#pragma unroll
  for (int d = 0; d < DHEAD; ++d) op[d] = f2bf(acc[d] * inv);
}

// ---------------------------------------------------------------------------
// Fused residual + LayerNorm over dim 128. One wave per token, 4 elems/lane.
// h <- LN(h + r) * g + b   (in place), plus bf16 copy.
// ---------------------------------------------------------------------------
__global__ void add_ln_kernel(float* __restrict__ h, const float* __restrict__ r,
                              const float* __restrict__ g, const float* __restrict__ be,
                              bf16* __restrict__ hb, int tokens)
{
  int wave = threadIdx.x >> 5;
  int lane = threadIdx.x & 31;
  int t = blockIdx.x * 4 + wave;
  if (t >= tokens) return;
  float* hp = h + (long long)t * HID;
  const float* rp = r + (long long)t * HID;

  float v[4]; float s = 0.0f;
#pragma unroll
  for (int i = 0; i < 4; ++i) { int d = lane + 32 * i; v[i] = hp[d] + rp[d]; s += v[i]; }
#pragma unroll
  for (int o = 16; o > 0; o >>= 1) s += __shfl_xor(s, o, 32);
  float mean = s * (1.0f / HID);
  float vs = 0.0f;
#pragma unroll
  for (int i = 0; i < 4; ++i) { float dd = v[i] - mean; vs += dd * dd; }
#pragma unroll
  for (int o = 16; o > 0; o >>= 1) vs += __shfl_xor(vs, o, 32);
  float inv = rsqrtf(vs * (1.0f / HID) + 1e-5f);
#pragma unroll
  for (int i = 0; i < 4; ++i) {
    int d = lane + 32 * i;
    float o = (v[i] - mean) * inv * g[d] + be[d];
    hp[d] = o;
    hb[(long long)t * HID + d] = f2bf(o);
  }
}

// ---------------------------------------------------------------------------
// graph = mean over nodes of ne
// ---------------------------------------------------------------------------
__global__ void graph_mean_kernel(const float* __restrict__ ne, float* __restrict__ graph)
{
  int idx = blockIdx.x * blockDim.x + threadIdx.x;
  if (idx >= NBATCH * HID) return;
  int b = idx / HID, d = idx % HID;
  float s = 0.0f;
  for (int n = 0; n < NNODE; ++n) s += ne[((long long)b * NNODE + n) * HID + d];
  graph[idx] = s * (1.0f / NNODE);
}

// ---------------------------------------------------------------------------
// Wcomb = wq @ qp_w (128x384, bf16) ; bcomb = wq @ qp_b  (one-time, tiny)
// ---------------------------------------------------------------------------
__global__ void wcomb_kernel(const float* __restrict__ wq, const float* __restrict__ qpw,
                             const float* __restrict__ qpb,
                             bf16* __restrict__ Wcb, float* __restrict__ bcomb)
{
  int idx = blockIdx.x * blockDim.x + threadIdx.x;
  if (idx >= HID * QDIM) return;
  int i = idx / QDIM, j = idx % QDIM;
  float s = 0.0f;
  for (int k = 0; k < HID; ++k) s += wq[i * HID + k] * qpw[k * QDIM + j];
  Wcb[idx] = f2bf(s);
  if (j == 0) {
    float t = 0.0f;
    for (int k = 0; k < HID; ++k) t += wq[i * HID + k] * qpb[k];
    bcomb[i] = t;
  }
}

__global__ void stopk_kernel(const float* __restrict__ wk, const float* __restrict__ stop,
                             float* __restrict__ stopk)
{
  int i = threadIdx.x;
  if (i >= HID) return;
  float s = 0.0f;
  for (int j = 0; j < HID; ++j) s += wk[i * HID + j] * stop[j];
  stopk[i] = s;
}

// ---------------------------------------------------------------------------
// Decoder state init: mask0 = (n==start), cur0 = start, fin0 = 0, tour[:,0]=start
// ---------------------------------------------------------------------------
__global__ void decode_init_kernel(int* __restrict__ cur, int* __restrict__ fin,
                                   unsigned char* __restrict__ mask, int* __restrict__ tours)
{
  int idx = blockIdx.x * blockDim.x + threadIdx.x;
  if (idx >= TOK * NNODE) return;
  int p = idx / NNODE, n = idx % NNODE;
  int b = p / NNODE, s = p % NNODE;
  mask[idx] = (n == s) ? 1 : 0;
  if (n == 0) {
    cur[p] = s;
    fin[p] = 0;
    tours[(long long)b * NNODE * (NNODE + 1) + (long long)s * (NNODE + 1)] = s;
  }
}

// Qin[p][0:128]=graph[b], [128:256]=start_e (constant across steps)
__global__ void qin_prefill_kernel(const float* __restrict__ ne, const float* __restrict__ graph,
                                   bf16* __restrict__ Qin)
{
  int idx = blockIdx.x * blockDim.x + threadIdx.x;
  if (idx >= TOK * 256) return;
  int p = idx / 256, d = idx % 256;
  int b = p / NNODE, s = p % NNODE;
  float v = (d < HID) ? graph[b * HID + d]
                      : ne[((long long)b * NNODE + s) * HID + (d - HID)];
  Qin[(long long)p * QDIM + d] = f2bf(v);
}

// Qin[p][256:384] = ne[b, cur[p]]  (per step)
__global__ void qin_update_kernel(const float* __restrict__ ne, const int* __restrict__ cur,
                                  bf16* __restrict__ Qin)
{
  int idx = blockIdx.x * blockDim.x + threadIdx.x;
  if (idx >= TOK * HID) return;
  int p = idx / HID, d = idx % HID;
  int b = p / NNODE;
  int c = cur[p];
  Qin[(long long)p * QDIM + 256 + d] = f2bf(ne[((long long)b * NNODE + c) * HID + d]);
}

// ---------------------------------------------------------------------------
// Masked argmax over [compat(200), stop] + state update. One 128-thread block
// per (batch,start) pair. Ties keep the lowest index (jnp.argmax semantics);
// stop (index 200) wins only on strict >.
// ---------------------------------------------------------------------------
__global__ void argmax_update_kernel(const float* __restrict__ compat,
                                     const float* __restrict__ qf,
                                     const float* __restrict__ stopk,
                                     unsigned char* __restrict__ mask,
                                     int* __restrict__ cur, int* __restrict__ fin,
                                     int* __restrict__ tours, int step)
{
  const int p = blockIdx.x;         // b*NNODE + s
  const int tid = threadIdx.x;      // 0..127
  const int b = p / NNODE, s = p % NNODE;
  __shared__ float sval[128];
  __shared__ int   sidx[128];

  // stop logit: dot(q[p], stop_k) * scale
  sval[tid] = qf[(long long)p * HID + tid] * stopk[tid];
  __syncthreads();
  for (int o = 64; o > 0; o >>= 1) { if (tid < o) sval[tid] += sval[tid + o]; __syncthreads(); }
  const float stop_l = sval[0] * CSCALE;
  __syncthreads();

  // node argmax (masked)
  const float* cp = compat + (long long)b * NNODE * NNODE + (long long)s * NNODE;
  const unsigned char* mp = mask + (long long)b * NNODE * NNODE + (long long)s * NNODE;
  float bv = -3.4e38f; int bi = 0x7fffffff;
  for (int n = tid; n < NNODE; n += 128) {
    float v = mp[n] ? NEGV : cp[n];
    if (v > bv || (v == bv && n < bi)) { bv = v; bi = n; }
  }
  sval[tid] = bv; sidx[tid] = bi;
  __syncthreads();
  for (int o = 64; o > 0; o >>= 1) {
    if (tid < o) {
      float v2 = sval[tid + o]; int i2 = sidx[tid + o];
      if (v2 > sval[tid] || (v2 == sval[tid] && i2 < sidx[tid])) { sval[tid] = v2; sidx[tid] = i2; }
    }
    __syncthreads();
  }

  if (tid == 0) {
    const float bnv = sval[0]; const int bni = sidx[0];
    const int nxt = (stop_l > bnv) ? NNODE : bni;
    const int is_stop = (nxt == NNODE);
    const int c = cur[p];
    const int nc = is_stop ? c : nxt;
    const int f = fin[p];
    if (!f && !is_stop) mask[(long long)b * NNODE * NNODE + (long long)s * NNODE + nc] = 1;
    cur[p] = nc;
    fin[p] = f | is_stop;
    tours[(long long)b * NNODE * (NNODE + 1) + (long long)s * (NNODE + 1) + 1 + step] = nc;
  }
}

// ---------------------------------------------------------------------------
// Host side
// ---------------------------------------------------------------------------
static inline void launch_gemm(hipStream_t st, const bf16* A, const bf16* B, const float* bias,
                               float* Cf, bf16* Cb, int M, int N, int K,
                               float alpha = 1.0f, int relu = 0,
                               int batch = 1, long long sA = 0, long long sB = 0, long long sC = 0)
{
  // block = 4 waves (2x2), each wave a 32x32 C tile -> block covers 64x64
  dim3 grid((N + 63) / 64, (M + 63) / 64, batch);
  wmma_gemm_kernel<<<grid, dim3(128), 0, st>>>(A, B, bias, Cf, Cb, M, N, K, alpha, relu, sA, sB, sC);
}

static inline void launch_cvt(hipStream_t st, const float* s, bf16* d, long long n)
{
  cvt_bf16_kernel<<<(unsigned)((n + 255) / 256), 256, 0, st>>>(s, d, n);
}

extern "C" void kernel_launch(void* const* d_in, const int* in_sizes, int n_in,
                              void* d_out, int out_size, void* d_ws, size_t ws_size,
                              hipStream_t stream)
{
  (void)in_sizes; (void)n_in; (void)out_size; (void)ws_size;

  // setup_inputs() dict order (recursive insertion order)
  const float* x       = (const float*)d_in[0];
  const float* embed_w = (const float*)d_in[1];
  const float* embed_b = (const float*)d_in[2];
  const float* in_w    = (const float*)d_in[3];   // (6,384,128)
  const float* in_b    = (const float*)d_in[4];   // (6,384)
  const float* out_w   = (const float*)d_in[5];   // (6,128,128)
  const float* out_b   = (const float*)d_in[6];   // (6,128)
  const float* l1_w    = (const float*)d_in[7];   // (6,2048,128)
  const float* l1_b    = (const float*)d_in[8];   // (6,2048)
  const float* l2_w    = (const float*)d_in[9];   // (6,128,2048)
  const float* l2_b    = (const float*)d_in[10];  // (6,128)
  const float* n1_w    = (const float*)d_in[11];
  const float* n1_b    = (const float*)d_in[12];
  const float* n2_w    = (const float*)d_in[13];
  const float* n2_b    = (const float*)d_in[14];
  const float* qp_w    = (const float*)d_in[15];  // (128,384)
  const float* qp_b    = (const float*)d_in[16];
  const float* wq      = (const float*)d_in[17];  // (128,128)
  const float* wk      = (const float*)d_in[18];  // (128,128)
  const float* stop    = (const float*)d_in[19];  // (128,)

  int* tours = (int*)d_out;                       // (64, 200, 201) int32

  // ---- workspace arena ----
  char* w = (char*)d_ws;
  size_t off = 0;
  auto alloc = [&](size_t bytes) -> void* {
    void* p = (void*)(w + off);
    off += (bytes + 255) & ~(size_t)255;
    return p;
  };
  float* h      = (float*)alloc((size_t)TOK * HID * 4);
  bf16*  hb     = (bf16*) alloc((size_t)TOK * HID * 2);
  float* qkvf   = (float*)alloc((size_t)TOK * QDIM * 4);
  bf16*  attnb  = (bf16*) alloc((size_t)TOK * HID * 2);
  float* tmpf   = (float*)alloc((size_t)TOK * HID * 4);
  bf16*  ff1b   = (bf16*) alloc((size_t)TOK * DFF * 2);
  bf16*  in_wb  = (bf16*) alloc((size_t)NLAYER * QDIM * HID * 2);
  bf16*  out_wb = (bf16*) alloc((size_t)NLAYER * HID * HID * 2);
  bf16*  l1_wb  = (bf16*) alloc((size_t)NLAYER * DFF * HID * 2);
  bf16*  l2_wb  = (bf16*) alloc((size_t)NLAYER * HID * DFF * 2);
  bf16*  wkb    = (bf16*) alloc((size_t)HID * HID * 2);
  bf16*  Wcb    = (bf16*) alloc((size_t)HID * QDIM * 2);
  float* bcomb  = (float*)alloc((size_t)HID * 4);
  float* graphf = (float*)alloc((size_t)NBATCH * HID * 4);
  bf16*  kb     = (bf16*) alloc((size_t)TOK * HID * 2);
  float* stopk  = (float*)alloc((size_t)HID * 4);
  bf16*  Qin    = (bf16*) alloc((size_t)TOK * QDIM * 2);
  float* qf     = (float*)alloc((size_t)TOK * HID * 4);
  bf16*  qb2    = (bf16*) alloc((size_t)TOK * HID * 2);
  float* compat = (float*)alloc((size_t)NBATCH * NNODE * NNODE * 4);
  int*   cur    = (int*)  alloc((size_t)TOK * 4);
  int*   fin    = (int*)  alloc((size_t)TOK * 4);
  unsigned char* mask = (unsigned char*)alloc((size_t)NBATCH * NNODE * NNODE);

  // ---- one-time weight conversion to bf16 ----
  launch_cvt(stream, in_w,  in_wb,  (long long)NLAYER * QDIM * HID);
  launch_cvt(stream, out_w, out_wb, (long long)NLAYER * HID * HID);
  launch_cvt(stream, l1_w,  l1_wb,  (long long)NLAYER * DFF * HID);
  launch_cvt(stream, l2_w,  l2_wb,  (long long)NLAYER * HID * DFF);
  launch_cvt(stream, wk,    wkb,    (long long)HID * HID);
  wcomb_kernel<<<(HID * QDIM + 255) / 256, 256, 0, stream>>>(wq, qp_w, qp_b, Wcb, bcomb);
  stopk_kernel<<<1, 128, 0, stream>>>(wk, stop, stopk);

  // ---- encoder ----
  embed_kernel<<<(TOK * HID + 255) / 256, 256, 0, stream>>>(x, embed_w, embed_b, h);
  launch_cvt(stream, h, hb, (long long)TOK * HID);

  for (int L = 0; L < NLAYER; ++L) {
    // qkv = h @ in_w^T + in_b
    launch_gemm(stream, hb, in_wb + (size_t)L * QDIM * HID, in_b + (size_t)L * QDIM,
                qkvf, nullptr, TOK, QDIM, HID);
    // attention (scalar flash-style) -> bf16
    attn_kernel<<<(NBATCH * NHEAD * NNODE + 255) / 256, 256, 0, stream>>>(qkvf, attnb);
    // o = attn @ out_w^T + out_b
    launch_gemm(stream, attnb, out_wb + (size_t)L * HID * HID, out_b + (size_t)L * HID,
                tmpf, nullptr, TOK, HID, HID);
    // h = LN(h + o)
    add_ln_kernel<<<TOK / 4, 128, 0, stream>>>(h, tmpf, n1_w + (size_t)L * HID,
                                               n1_b + (size_t)L * HID, hb, TOK);
    // f1 = relu(h @ l1_w^T + l1_b)   (bf16 only)
    launch_gemm(stream, hb, l1_wb + (size_t)L * DFF * HID, l1_b + (size_t)L * DFF,
                nullptr, ff1b, TOK, DFF, HID, 1.0f, /*relu=*/1);
    // f2 = f1 @ l2_w^T + l2_b
    launch_gemm(stream, ff1b, l2_wb + (size_t)L * HID * DFF, l2_b + (size_t)L * HID,
                tmpf, nullptr, TOK, HID, DFF);
    // h = LN(h + f2)
    add_ln_kernel<<<TOK / 4, 128, 0, stream>>>(h, tmpf, n2_w + (size_t)L * HID,
                                               n2_b + (size_t)L * HID, hb, TOK);
  }

  // ---- decoder setup ----
  graph_mean_kernel<<<(NBATCH * HID + 255) / 256, 256, 0, stream>>>(h, graphf);
  // k_nodes = ne @ wk^T (bf16 only, consumed by compat GEMM)
  launch_gemm(stream, hb, wkb, nullptr, nullptr, kb, TOK, HID, HID);
  qin_prefill_kernel<<<(TOK * 256 + 255) / 256, 256, 0, stream>>>(h, graphf, Qin);
  decode_init_kernel<<<(TOK * NNODE + 255) / 256, 256, 0, stream>>>(cur, fin, mask, tours);

  // ---- greedy decode: 200 sequential steps ----
  for (int t = 0; t < NNODE; ++t) {
    // cur_e slice of query input (cur0 == start at t=0, so uniform loop is correct)
    qin_update_kernel<<<(TOK * HID + 255) / 256, 256, 0, stream>>>(h, cur, Qin);
    // q = Qin @ Wcomb^T + bcomb   (folds qp_w and wq projections)
    launch_gemm(stream, Qin, Wcb, bcomb, qf, qb2, TOK, HID, QDIM);
    // compat[b] = scale * q_b (200x128) @ k_nodes_b (200x128)^T, batched over 64
    launch_gemm(stream, qb2, kb, nullptr, compat, nullptr, NNODE, NNODE, HID,
                CSCALE, 0, NBATCH,
                (long long)NNODE * HID, (long long)NNODE * HID, (long long)NNODE * NNODE);
    // masked argmax over [compat, stop] + state update + tour write
    argmax_update_kernel<<<TOK, 128, 0, stream>>>(compat, qf, stopk, mask, cur, fin, tours, t);
  }
}
